// HGRNBitMLP_3908420240253
// MI455X (gfx1250) — 8-bit matrix operand order repaired and checked by reading, compiled, not run on hardware
//
#include <hip/hip_runtime.h>
#include <hip/hip_bf16.h>
#include <stdint.h>


typedef int v8i __attribute__((ext_vector_type(8)));
typedef int v4i __attribute__((ext_vector_type(4)));

#if defined(__AMDGCN__) && __has_builtin(__builtin_amdgcn_global_load_async_to_lds_b128)
#define HAVE_ASYNC_LDS 1
#else
#define HAVE_ASYNC_LDS 0
#endif

#if HAVE_ASYNC_LDS
typedef int v4i_vs __attribute__((vector_size(16)));
typedef __attribute__((address_space(1))) v4i_vs v4i_gas;
typedef __attribute__((address_space(3))) v4i_vs v4i_las;
#endif

__device__ __forceinline__ void copy16_g2l(const int8_t* g, int8_t* l) {
#if HAVE_ASYNC_LDS
  __builtin_amdgcn_global_load_async_to_lds_b128(
      (v4i_gas*)(uintptr_t)g, (v4i_las*)l, 0, 0);
#else
  *(v4i*)l = *(const v4i*)g;
#endif
}

__device__ __forceinline__ void wait_async_lds() {
#if defined(__AMDGCN__) && __has_builtin(__builtin_amdgcn_s_wait_asynccnt)
  __builtin_amdgcn_s_wait_asynccnt(0);
#elif HAVE_ASYNC_LDS
  asm volatile("s_wait_asynccnt 0x0" ::: "memory");
#endif
}

__global__ __launch_bounds__(256) void abs_partial_kernel(
    const float* __restrict__ w, long long n, float* __restrict__ partial) {
  __shared__ float red[256];
  long long i = blockIdx.x * 256LL + threadIdx.x;
  const long long stride = (long long)gridDim.x * 256LL;
  float s = 0.f;
  for (; i < n; i += stride) s += fabsf(w[i]);
  red[threadIdx.x] = s;
  __syncthreads();
  for (int t = 128; t > 0; t >>= 1) {
    if ((int)threadIdx.x < t) red[threadIdx.x] += red[threadIdx.x + t];
    __syncthreads();
  }
  if (threadIdx.x == 0) partial[blockIdx.x] = red[0];
}

__global__ __launch_bounds__(256) void abs_final_kernel(
    const float* __restrict__ partial, int np, float* __restrict__ total) {
  __shared__ float red[256];
  float s = 0.f;
  for (int i = threadIdx.x; i < np; i += 256) s += partial[i];
  red[threadIdx.x] = s;
  __syncthreads();
  for (int t = 128; t > 0; t >>= 1) {
    if ((int)threadIdx.x < t) red[threadIdx.x] += red[threadIdx.x + t];
    __syncthreads();
  }
  if (threadIdx.x == 0) total[0] = red[0];
}

__global__ __launch_bounds__(256) void weight_quant_kernel(
    const float* __restrict__ w, int8_t* __restrict__ q, long long n,
    const float* __restrict__ abssum, float inv_n) {
  const float mean = fmaxf(abssum[0] * inv_n, 1e-5f);
  const float sc = 1.f / mean;
  long long i = blockIdx.x * 256LL + threadIdx.x;
  const long long stride = (long long)gridDim.x * 256LL;
  for (; i < n; i += stride) {
    float r = rintf(w[i] * sc);
    r = fminf(fmaxf(r, -1.f), 1.f);
    q[i] = (int8_t)r;
  }
}

__global__ __launch_bounds__(256) void act_rmsnorm_quant_kernel(
    const float* __restrict__ x, const float* __restrict__ g,
    int8_t* __restrict__ q, float* __restrict__ scale_inv, int L) {
  __shared__ float red[256];
  const int row = blockIdx.x;
  const int tid = threadIdx.x;
  const float* xr = x + (size_t)row * L;
  const int n = L >> 8;
  float v[24];
  float ss = 0.f;
  for (int i = 0; i < n; ++i) {
    float xv = xr[tid + (i << 8)];
    v[i] = xv;
    ss += xv * xv;
  }
  red[tid] = ss;
  __syncthreads();
  for (int t = 128; t > 0; t >>= 1) {
    if (tid < t) red[tid] += red[tid + t];
    __syncthreads();
  }
  const float rs = rsqrtf(red[0] / (float)L + 1e-6f);
  __syncthreads();
  float amax = 0.f;
  for (int i = 0; i < n; ++i) {
    float xn = v[i] * rs * g[tid + (i << 8)];
    v[i] = xn;
    amax = fmaxf(amax, fabsf(xn));
  }
  red[tid] = amax;
  __syncthreads();
  for (int t = 128; t > 0; t >>= 1) {
    if (tid < t) red[tid] = fmaxf(red[tid], red[tid + t]);
    __syncthreads();
  }
  const float cmax = fmaxf(red[0], 1e-5f);
  const float sc = 127.f / cmax;
  int8_t* qr = q + (size_t)row * L;
  for (int i = 0; i < n; ++i) {
    float r = rintf(v[i] * sc);
    r = fminf(fmaxf(r, -128.f), 127.f);
    qr[tid + (i << 8)] = (int8_t)r;
  }
  if (tid == 0) scale_inv[row] = cmax * (1.f / 127.f);
}

template <bool FUSE>
__global__ __launch_bounds__(256) void bitnet_gemm_kernel(
    const int8_t* __restrict__ A, const int8_t* __restrict__ W,
    const float* __restrict__ ascale, const float* __restrict__ abssum,
    float winv_n, float* __restrict__ out, int M, int K, int N) {
  __shared__ int8_t sA[2][128 * 64];
  __shared__ int8_t sB[2][128 * 64];
  __shared__ int8_t sB2[2][128 * 64];

  const int tid  = threadIdx.x;
  const int lane = tid & 31;
  const int wid  = tid >> 5;
  const int wm   = wid & 3;
  const int wn   = wid >> 2;
  const int half = lane >> 4;
  const int l15  = lane & 15;
  const int bN0  = blockIdx.x * 128;
  const int bM0  = blockIdx.y * 128;

  const int idx0 = tid, idx1 = tid + 256;
  const int r0 = idx0 >> 2, c0 = (idx0 & 3) << 4;
  const int r1 = idx1 >> 2, c1 = (idx1 & 3) << 4;

  v8i acc[2][4] = {};
  v8i acc2[2][4] = {};

  auto stage = [&](int buf, int k0) {
#if HAVE_ASYNC_LDS
    copy16_g2l(&A[(size_t)(bM0 + r0) * K + k0 + c0], &sA[buf][r0 * 64 + c0]);
    copy16_g2l(&A[(size_t)(bM0 + r1) * K + k0 + c1], &sA[buf][r1 * 64 + c1]);
    copy16_g2l(&W[(size_t)(bN0 + r0) * K + k0 + c0], &sB[buf][r0 * 64 + c0]);
    copy16_g2l(&W[(size_t)(bN0 + r1) * K + k0 + c1], &sB[buf][r1 * 64 + c1]);
    if (FUSE) {
      copy16_g2l(&W[(size_t)(N + bN0 + r0) * K + k0 + c0],
                 &sB2[buf][r0 * 64 + c0]);
      copy16_g2l(&W[(size_t)(N + bN0 + r1) * K + k0 + c1],
                 &sB2[buf][r1 * 64 + c1]);
    }
#else
    v4i tA0 = *(const v4i*)&A[(size_t)(bM0 + r0) * K + k0 + c0];
    v4i tA1 = *(const v4i*)&A[(size_t)(bM0 + r1) * K + k0 + c1];
    v4i tB0 = *(const v4i*)&W[(size_t)(bN0 + r0) * K + k0 + c0];
    v4i tB1 = *(const v4i*)&W[(size_t)(bN0 + r1) * K + k0 + c1];
    v4i tC0 = {}, tC1 = {};
    if (FUSE) {
      tC0 = *(const v4i*)&W[(size_t)(N + bN0 + r0) * K + k0 + c0];
      tC1 = *(const v4i*)&W[(size_t)(N + bN0 + r1) * K + k0 + c1];
    }
    *(v4i*)&sA[buf][r0 * 64 + c0] = tA0;
    *(v4i*)&sA[buf][r1 * 64 + c1] = tA1;
    *(v4i*)&sB[buf][r0 * 64 + c0] = tB0;
    *(v4i*)&sB[buf][r1 * 64 + c1] = tB1;
    if (FUSE) {
      *(v4i*)&sB2[buf][r0 * 64 + c0] = tC0;
      *(v4i*)&sB2[buf][r1 * 64 + c1] = tC1;
    }
#endif
  };

  stage(0, 0);
  int buf = 0;
  for (int k0 = 0; k0 < K; k0 += 64) {
    wait_async_lds();
    __syncthreads();
    if (k0 + 64 < K) stage(buf ^ 1, k0 + 64);

    v8i afrag[2];
#pragma unroll
    for (int mf = 0; mf < 2; ++mf) {
      const int row = wm * 32 + mf * 16 + l15;
#pragma unroll
      for (int v = 0; v < 8; ++v) {
        const int kk = ((v >> 2) << 5) + (half << 4) + ((v & 3) << 2);
        afrag[mf][v] = *(const int*)&sA[buf][row * 64 + kk];
      }
    }
    v8i bfrag[4];
#pragma unroll
    for (int nf = 0; nf < 4; ++nf) {
      const int col = wn * 64 + nf * 16 + l15;
#pragma unroll
      for (int v = 0; v < 8; ++v) {
        const int kk = ((v >> 2) << 5) + (half << 4) + ((v & 3) << 2);
        bfrag[nf][v] = *(const int*)&sB[buf][col * 64 + kk];
      }
    }
#pragma unroll
    for (int mf = 0; mf < 2; ++mf)
#pragma unroll
      for (int nf = 0; nf < 4; ++nf)
        acc[mf][nf] = __builtin_amdgcn_wmma_i32_16x16x64_iu8(
            true, afrag[mf], true, bfrag[nf], acc[mf][nf], false, false);

    if (FUSE) {
#pragma unroll
      for (int nf = 0; nf < 4; ++nf) {
        const int col = wn * 64 + nf * 16 + l15;
#pragma unroll
        for (int v = 0; v < 8; ++v) {
          const int kk = ((v >> 2) << 5) + (half << 4) + ((v & 3) << 2);
          bfrag[nf][v] = *(const int*)&sB2[buf][col * 64 + kk];
        }
      }
#pragma unroll
      for (int mf = 0; mf < 2; ++mf)
#pragma unroll
        for (int nf = 0; nf < 4; ++nf)
          acc2[mf][nf] = __builtin_amdgcn_wmma_i32_16x16x64_iu8(
              true, afrag[mf], true, bfrag[nf], acc2[mf][nf], false, false);
    }
    buf ^= 1;
  }

  const float wmean = fmaxf(abssum[0] * winv_n, 1e-5f);
#pragma unroll
  for (int mf = 0; mf < 2; ++mf) {
#pragma unroll
    for (int nf = 0; nf < 4; ++nf) {
      const int colg = bN0 + wn * 64 + nf * 16 + l15;
#pragma unroll
      for (int r = 0; r < 8; ++r) {
        const int rowg = bM0 + wm * 32 + mf * 16 + r + (half << 3);
        const float dq = ascale[rowg] * wmean;
        const float gv = (float)acc[mf][nf][r] * dq;
        if (FUSE) {
          const float uv = (float)acc2[mf][nf][r] * dq;
          const float si = gv / (1.f + __expf(-gv));
          out[(size_t)rowg * N + colg] = si * uv;
        } else {
          out[(size_t)rowg * N + colg] = gv;
        }
      }
    }
  }
}

extern "C" void kernel_launch(void* const* d_in, const int* in_sizes, int n_in,
                              void* d_out, int out_size, void* d_ws,
                              size_t ws_size, hipStream_t stream) {
  const float* x      = (const float*)d_in[0];
  const float* w_gate = (const float*)d_in[1];
  const float* g_gate = (const float*)d_in[2];
  const float* w_down = (const float*)d_in[3];
  const float* g_down = (const float*)d_in[4];
  float* out = (float*)d_out;

  const int hidden = in_sizes[2];
  const int inter  = in_sizes[4];
  const int M      = in_sizes[0] / hidden;
  const long long n_wg = (long long)2 * inter * hidden;
  const long long n_wd = (long long)hidden * inter;

  size_t off = 0;
  auto carve = [&](size_t bytes) {
    size_t o = off;
    off = (off + bytes + 255) & ~(size_t)255;
    return o;
  };
  char* ws = (char*)d_ws;
  int8_t* wq_gate = (int8_t*)(ws + carve((size_t)n_wg));
  int8_t* wq_down = (int8_t*)(ws + carve((size_t)n_wd));
  int8_t* xq      = (int8_t*)(ws + carve((size_t)M * hidden));
  int8_t* hq      = (int8_t*)(ws + carve((size_t)M * inter));
  float*  xscale  = (float*)(ws + carve((size_t)M * 4));
  float*  hscale  = (float*)(ws + carve((size_t)M * 4));
  float*  h       = (float*)(ws + carve((size_t)M * inter * 4));
  float*  part_g  = (float*)(ws + carve(1024 * 4));
  float*  part_d  = (float*)(ws + carve(1024 * 4));
  float*  sum_g   = (float*)(ws + carve(4));
  float*  sum_d   = (float*)(ws + carve(4));

  abs_partial_kernel<<<1024, 256, 0, stream>>>(w_gate, n_wg, part_g);
  abs_partial_kernel<<<1024, 256, 0, stream>>>(w_down, n_wd, part_d);
  abs_final_kernel<<<1, 256, 0, stream>>>(part_g, 1024, sum_g);
  abs_final_kernel<<<1, 256, 0, stream>>>(part_d, 1024, sum_d);
  weight_quant_kernel<<<2048, 256, 0, stream>>>(w_gate, wq_gate, n_wg, sum_g,
                                                1.f / (float)n_wg);
  weight_quant_kernel<<<2048, 256, 0, stream>>>(w_down, wq_down, n_wd, sum_d,
                                                1.f / (float)n_wd);

  act_rmsnorm_quant_kernel<<<M, 256, 0, stream>>>(x, g_gate, xq, xscale,
                                                  hidden);

  bitnet_gemm_kernel<true><<<dim3(inter / 128, M / 128), 256, 0, stream>>>(
      xq, wq_gate, xscale, sum_g, 1.f / (float)n_wg, h, M, hidden, inter);

  act_rmsnorm_quant_kernel<<<M, 256, 0, stream>>>(h, g_down, hq, hscale,
                                                  inter);

  bitnet_gemm_kernel<false><<<dim3(hidden / 128, M / 128), 256, 0, stream>>>(
      hq, wq_down, hscale, sum_d, 1.f / (float)n_wd, out, M, inter, hidden);
}
